// BlockwiseParallelTransformerAttention_14697378087021
// MI455X (gfx1250) — matrix operand order repaired and checked by reading, not compiled, not run on hardware
//
#include <hip/hip_runtime.h>
#include <hip/hip_bf16.h>

typedef __attribute__((ext_vector_type(16))) _Float16 v16h;
typedef __attribute__((ext_vector_type(8)))  float    v8f;

#define B_   4
#define S_   2048
#define D_   512
#define HID_ 64
#define L_   6
#define EPS_ 1e-5f

// ---------------------------------------------------------------------------
// WMMA helpers (CDNA5 wave32, 16x16x32 f16 -> f32)
// ---------------------------------------------------------------------------
__device__ __forceinline__ v8f wmma_f16(v16h a, v16h b, v8f c) {
    return __builtin_amdgcn_wmma_f32_16x16x32_f16(
        /*neg_a=*/false, a, /*neg_b=*/false, b,
        /*c_mod=*/(short)0, c, /*reuse_a=*/false, /*reuse_b=*/false);
}

// CDNA5 async global->LDS copy (16B per lane), tracked by ASYNCcnt.
// VDST = LDS byte address (low 32 bits of generic LDS pointer), VADDR = 64-bit
// global address.  Syntax per CDNA5 ISA / gfx1250 VFLAT.
__device__ __forceinline__ void async_copy16(void* ldsDst, const void* gSrc) {
    unsigned lds = (unsigned)(unsigned long long)ldsDst;
    asm volatile("global_load_async_to_lds_b128 %0, %1, off"
                 :: "v"(lds), "v"(gSrc) : "memory");
}
__device__ __forceinline__ void wait_async0() {
    asm volatile("s_wait_asynccnt 0x0" ::: "memory");
}

// A-matrix fragment (16x32, f16, row-major buffer with row stride ld halves).
// lanes0-15 M=lane K-base 0; lanes16-31 M=lane-16 K-base 16.
// elements 0..7  -> K = base+0..7   (contiguous 16B)
// elements 8..15 -> K = base+8..15  (contiguous 16B)
__device__ __forceinline__ v16h loadAfrag(const _Float16* base, int ld, int kBase) {
    int lane = threadIdx.x & 31;
    int m    = lane & 15;
    int ko   = (lane & 16) ? 16 : 0;
    const _Float16* p0 = base + (size_t)m * ld + kBase + ko;
    v16h a;
    ((float4*)&a)[0] = *(const float4*)p0;         // K = base..base+7
    ((float4*)&a)[1] = *(const float4*)(p0 + 8);   // K = base+8..base+15
    return a;
}

// B-matrix fragment (32x16, f16) from a K-transposed buffer bt[N][K]
// (row stride ldk halves). lane -> column N = colBase+(lane&15);
// lanes0-15 carry K = kBase+0..15, lanes16-31 carry K = kBase+16..31.
__device__ __forceinline__ v16h loadBfragT(const _Float16* bt, int ldk,
                                           int colBase, int kBase) {
    int lane = threadIdx.x & 31;
    int col  = colBase + (lane & 15);
    int ko   = (lane & 16) ? 16 : 0;
    const _Float16* p = bt + (size_t)col * ldk + kBase + ko;
    v16h b;
    ((float4*)&b)[0] = *(const float4*)p;        // K = kBase+ko .. +7
    ((float4*)&b)[1] = *(const float4*)(p + 8);  // K = kBase+ko+8 .. +15
    return b;
}

// ---------------------------------------------------------------------------
// Prep kernels
// ---------------------------------------------------------------------------
__global__ void convert_x_kernel(const float* __restrict__ x,
                                 _Float16* __restrict__ x16,
                                 _Float16* __restrict__ h16,
                                 float* __restrict__ h32, int n) {
    int i = blockIdx.x * blockDim.x + threadIdx.x;
    if (i < n) {
        float v = x[i];
        x16[i] = (_Float16)v;
        h16[i] = (_Float16)v;
        h32[i] = v;
    }
}

// src f32 [K][N]  ->  dst f16 [N][K]
__global__ void transpose_w_kernel(const float* __restrict__ src,
                                   _Float16* __restrict__ dst, int K, int N) {
    int i = blockIdx.x * blockDim.x + threadIdx.x;
    if (i < K * N) {
        int n = i / K, k = i % K;
        dst[i] = (_Float16)src[(size_t)k * N + n];
    }
}

// ---------------------------------------------------------------------------
// GEMM: out[M][N] = A[M][K] @ BT^T + bias   (A f16 row-major, BT f16 [N][K])
// storeMode 0: out row-major [M][N] f16
// storeMode 1: out transposed-per-batch f16 [B][N][S]  (for V^T)
// Block: 256 threads = 8 waves; tile 128(M) x 64(N).
// B staged through LDS with double-buffered async global->LDS copies.
// ---------------------------------------------------------------------------
__global__ __launch_bounds__(256) void gemm_bias_f16(
    const _Float16* __restrict__ A, const _Float16* __restrict__ BT,
    const float* __restrict__ bias, _Float16* __restrict__ out,
    int M, int N, int K, int storeMode) {
    __shared__ __align__(16) _Float16 Bst[2][64 * 32];

    int tid  = threadIdx.x;
    int wave = tid >> 5, lane = tid & 31;
    int tilesN = N >> 6;
    int tileN  = blockIdx.x % tilesN;
    int tileM  = blockIdx.x / tilesN;
    int m0 = tileM * 128 + wave * 16;
    int n0 = tileN * 64;

    // Per-thread 16B slice of the 64x32 B tile.
    int idx = tid * 8;
    int nl = idx >> 5, ko = idx & 31;
    const _Float16* gB = &BT[(size_t)(n0 + nl) * K + ko];
    _Float16* ldsB0 = &Bst[0][nl * 32 + ko];
    _Float16* ldsB1 = &Bst[1][nl * 32 + ko];

    async_copy16(ldsB0, gB);            // stage k0 = 0
    int cur = 0;
    v8f acc[4] = {};
    for (int k0 = 0; k0 < K; k0 += 32) {
        wait_async0();                  // my slice of stage(k0) landed
        __syncthreads();                // everyone's landed; prev buf free
        if (k0 + 32 < K)                // prefetch next tile into other buf
            async_copy16(cur ? ldsB0 : ldsB1, gB + k0 + 32);

        v16h a = loadAfrag(A + (size_t)m0 * K, K, k0);
        const _Float16* bb = Bst[cur];
        v16h bf[4];
#pragma unroll
        for (int t = 0; t < 4; ++t) bf[t] = loadBfragT(bb, 32, t * 16, 0);
#pragma unroll
        for (int t = 0; t < 4; ++t) acc[t] = wmma_f16(a, bf[t], acc[t]);
        cur ^= 1;
    }

    int hi = (lane & 16) ? 8 : 0;
#pragma unroll
    for (int t = 0; t < 4; ++t) {
        int col = n0 + t * 16 + (lane & 15);
        float bv = bias[col];
#pragma unroll
        for (int r = 0; r < 8; ++r) {
            int row = m0 + r + hi;
            float val = acc[t][r] + bv;
            if (storeMode == 0) {
                out[(size_t)row * N + col] = (_Float16)val;
            } else {
                int bb2 = row / S_, s = row % S_;
                out[((size_t)bb2 * N + col) * S_ + s] = (_Float16)val;
            }
        }
    }
}

// ---------------------------------------------------------------------------
// Fused flash-attention + FFN + residual + 2x LayerNorm.
// One block (8 waves) per (batch, 16-query-row tile).
// ---------------------------------------------------------------------------
__global__ __launch_bounds__(256) void attn_ffn_ln_kernel(
    const _Float16* __restrict__ q16,   // [B][S][D]
    const _Float16* __restrict__ k16,   // [B][S][D]
    const _Float16* __restrict__ vT16,  // [B][D][S]
    const _Float16* __restrict__ W1T,   // [HID][D]
    const float* __restrict__ b1,
    const _Float16* __restrict__ W2T,   // [D][HID]
    const float* __restrict__ b2,
    const float* __restrict__ ln1g, const float* __restrict__ ln1b,
    const float* __restrict__ ln2g, const float* __restrict__ ln2b,
    const float* __restrict__ hOld32,   // [B][S][D]
    float* __restrict__ hNew32,         // [B][S][D]
    _Float16* __restrict__ hNew16) {    // [B][S][D]
    __shared__ __align__(16) _Float16 qbuf[16 * 512];   // q tile / attn tile
    __shared__ __align__(16) float    sbuf[16 * 128];   // raw scores (chunk)
    __shared__ __align__(16) _Float16 pbuf[16 * 128];   // probs f16 (chunk)
    __shared__ __align__(16) _Float16 hidbuf[16 * 64];  // FFN hidden
    __shared__ float mrow[16], lrow[16], crow[16];
    __shared__ float red0[16], red1[16];
    __shared__ float mu1[16], rs1[16];

    int tid = threadIdx.x, wave = tid >> 5, lane = tid & 31;
    int b    = blockIdx.x >> 7;     // 128 row-tiles per batch
    int qb   = blockIdx.x & 127;
    int row0 = qb * 16;

    const _Float16* qbase = q16 + ((size_t)b * S_ + row0) * D_;
    const _Float16* kbase = k16 + (size_t)b * S_ * D_;
    const _Float16* vtb   = vT16 + (size_t)b * D_ * S_;

    // Async-load 16x512 q tile into LDS (4 x 16B per thread)
#pragma unroll
    for (int e = 0; e < 4; ++e)
        async_copy16(&qbuf[tid * 32 + e * 8], &qbase[tid * 32 + e * 8]);
    if (tid < 16) { mrow[tid] = -1e30f; lrow[tid] = 0.f; }
    wait_async0();
    __syncthreads();

    const float scale_attn = 0.125f;               // 1/sqrt(64)
    const float scale_out  = 0.04419417382415922f; // 1/sqrt(512)
    int hi = (lane & 16) ? 8 : 0;

    v8f acc[4] = {};   // this wave's 16 x 64 output slice (cols wave*64..)

    for (int chunk = 0; chunk < S_; chunk += 128) {
        // ---- scores: wave w handles keys [chunk + 16w, +16) --------------
        int keyBase = chunk + wave * 16;
        v8f sc = {};
#pragma unroll 4
        for (int k0 = 0; k0 < D_; k0 += 32) {
            v16h a  = loadAfrag(qbuf, D_, k0);
            v16h bb = loadBfragT(kbase, D_, keyBase, k0); // K=d contiguous
            sc = wmma_f16(a, bb, sc);
        }
#pragma unroll
        for (int r = 0; r < 8; ++r)
            sbuf[(r + hi) * 128 + wave * 16 + (lane & 15)] = sc[r] * scale_attn;
        __syncthreads();

        // ---- online softmax stats (one thread per row) -------------------
        if (tid < 16) {
            float mold = mrow[tid];
            float mx = mold;
            for (int j = 0; j < 128; ++j) mx = fmaxf(mx, sbuf[tid * 128 + j]);
            float corr = __expf(mold - mx);
            float sum = 0.f;
            for (int j = 0; j < 128; ++j) {
                float p = __expf(sbuf[tid * 128 + j] - mx);
                pbuf[tid * 128 + j] = (_Float16)p;
                sum += p;
            }
            mrow[tid] = mx;
            lrow[tid] = lrow[tid] * corr + sum;
            crow[tid] = corr;
        }
        __syncthreads();

        // ---- rescale running accumulators --------------------------------
#pragma unroll
        for (int r = 0; r < 8; ++r) {
            float f = crow[r + hi];
#pragma unroll
            for (int t = 0; t < 4; ++t) acc[t][r] *= f;
        }

        // ---- P @ V over this 128-key chunk -------------------------------
        for (int kk = 0; kk < 128; kk += 32) {
            v16h a = loadAfrag(pbuf, 128, kk);
            v16h bf[4];
#pragma unroll
            for (int t = 0; t < 4; ++t)
                bf[t] = loadBfragT(vtb, S_, wave * 64 + t * 16, chunk + kk);
#pragma unroll
            for (int t = 0; t < 4; ++t) acc[t] = wmma_f16(a, bf[t], acc[t]);
        }
        __syncthreads();
    }

    // ---- attn epilogue: normalize + scale_out, stash f16 tile in qbuf ----
#pragma unroll
    for (int r = 0; r < 8; ++r) {
        float inv = scale_out / lrow[r + hi];
#pragma unroll
        for (int t = 0; t < 4; ++t)
            qbuf[(r + hi) * 512 + wave * 64 + t * 16 + (lane & 15)] =
                (_Float16)(acc[t][r] * inv);
    }
    __syncthreads();

    // ---- FFN1: hid = relu(attn @ W1 + b1), 16x64 (waves 0..3) ------------
    if (wave < 4) {
        v8f h1 = {};
#pragma unroll 4
        for (int k0 = 0; k0 < D_; k0 += 32) {
            v16h a  = loadAfrag(qbuf, D_, k0);
            v16h bb = loadBfragT(W1T, D_, wave * 16, k0);
            h1 = wmma_f16(a, bb, h1);
        }
        int col = wave * 16 + (lane & 15);
        float bv = b1[col];
#pragma unroll
        for (int r = 0; r < 8; ++r) {
            float v = h1[r] + bv;
            hidbuf[(r + hi) * 64 + col] = (_Float16)(v > 0.f ? v : 0.f);
        }
    }
    if (tid < 16) { red0[tid] = 0.f; red1[tid] = 0.f; }
    __syncthreads();

    // ---- FFN2: ff = hid @ W2 + b2 (each wave 16x64 slice) ----------------
    v8f ff[4] = {};
#pragma unroll
    for (int k0 = 0; k0 < HID_; k0 += 32) {
        v16h a = loadAfrag(hidbuf, HID_, k0);
        v16h bf[4];
#pragma unroll
        for (int t = 0; t < 4; ++t)
            bf[t] = loadBfragT(W2T, HID_, wave * 64 + t * 16, k0);
#pragma unroll
        for (int t = 0; t < 4; ++t) ff[t] = wmma_f16(a, bf[t], ff[t]);
    }

    // ---- residual add + LN statistics (round 1) --------------------------
    const float* hOld = hOld32 + ((size_t)b * S_ + row0) * D_;
    float tval[4][8];
#pragma unroll
    for (int t = 0; t < 4; ++t) {
        int col = wave * 64 + t * 16 + (lane & 15);
        float bv = b2[col];
#pragma unroll
        for (int r = 0; r < 8; ++r) {
            int row = r + hi;
            float v = ff[t][r] + bv + hOld[(size_t)row * D_ + col];
            tval[t][r] = v;
            atomicAdd(&red0[row], v);
            atomicAdd(&red1[row], v * v);
        }
    }
    __syncthreads();
    if (tid < 16) {
        float mu  = red0[tid] * (1.f / (float)D_);
        float var = red1[tid] * (1.f / (float)D_) - mu * mu;
        mu1[tid] = mu;
        rs1[tid] = rsqrtf(var + EPS_);
        red0[tid] = 0.f;
        red1[tid] = 0.f;
    }
    __syncthreads();

    // ---- LN1 apply + stats for LN2 (round 2) -----------------------------
    float yval[4][8];
#pragma unroll
    for (int t = 0; t < 4; ++t) {
        int col = wave * 64 + t * 16 + (lane & 15);
        float g = ln1g[col], bb_ = ln1b[col];
#pragma unroll
        for (int r = 0; r < 8; ++r) {
            int row = r + hi;
            float y = (tval[t][r] - mu1[row]) * rs1[row] * g + bb_;
            yval[t][r] = y;
            atomicAdd(&red0[row], y);
            atomicAdd(&red1[row], y * y);
        }
    }
    __syncthreads();
    if (tid < 16) {
        float mu  = red0[tid] * (1.f / (float)D_);
        float var = red1[tid] * (1.f / (float)D_) - mu * mu;
        mu1[tid] = mu;
        rs1[tid] = rsqrtf(var + EPS_);
    }
    __syncthreads();

    // ---- LN2 apply + writeback -------------------------------------------
    float* outRow32 = hNew32 + ((size_t)b * S_ + row0) * D_;
    _Float16* outRow16 = hNew16 + ((size_t)b * S_ + row0) * D_;
#pragma unroll
    for (int t = 0; t < 4; ++t) {
        int col = wave * 64 + t * 16 + (lane & 15);
        float g = ln2g[col], bb_ = ln2b[col];
#pragma unroll
        for (int r = 0; r < 8; ++r) {
            int row = r + hi;
            float z = (yval[t][r] - mu1[row]) * rs1[row] * g + bb_;
            outRow32[(size_t)row * D_ + col] = z;
            outRow16[(size_t)row * D_ + col] = (_Float16)z;
        }
    }
}

// ---------------------------------------------------------------------------
// Host side
// ---------------------------------------------------------------------------
extern "C" void kernel_launch(void* const* d_in, const int* in_sizes, int n_in,
                              void* d_out, int out_size, void* d_ws, size_t ws_size,
                              hipStream_t stream) {
    const float* x    = (const float*)d_in[0];
    const float* Wq   = (const float*)d_in[1];
    const float* bq   = (const float*)d_in[2];
    const float* Wk   = (const float*)d_in[3];
    const float* bk   = (const float*)d_in[4];
    const float* Wv   = (const float*)d_in[5];
    const float* bv   = (const float*)d_in[6];
    const float* W1   = (const float*)d_in[7];
    const float* b1   = (const float*)d_in[8];
    const float* W2   = (const float*)d_in[9];
    const float* b2   = (const float*)d_in[10];
    const float* ln1g = (const float*)d_in[11];
    const float* ln1b = (const float*)d_in[12];
    const float* ln2g = (const float*)d_in[13];
    const float* ln2b = (const float*)d_in[14];

    const size_t BSD = (size_t)B_ * S_ * D_;
    char* ws = (char*)d_ws;
    auto bump = [&](size_t bytes) {
        void* p = (void*)ws;
        ws += (bytes + 255) & ~(size_t)255;
        return p;
    };
    float*    h32 = (float*)bump(BSD * 4);
    _Float16* h16 = (_Float16*)bump(BSD * 2);
    _Float16* x16 = (_Float16*)bump(BSD * 2);
    _Float16* q16 = (_Float16*)bump(BSD * 2);
    _Float16* k16 = (_Float16*)bump(BSD * 2);
    _Float16* vT  = (_Float16*)bump(BSD * 2);
    _Float16* WqT = (_Float16*)bump((size_t)D_ * D_ * 2);
    _Float16* WkT = (_Float16*)bump((size_t)D_ * D_ * 2);
    _Float16* WvT = (_Float16*)bump((size_t)D_ * D_ * 2);
    _Float16* W1T = (_Float16*)bump((size_t)D_ * HID_ * 2);
    _Float16* W2T = (_Float16*)bump((size_t)HID_ * D_ * 2);

    int n = (int)BSD;
    convert_x_kernel<<<(n + 255) / 256, 256, 0, stream>>>(x, x16, h16, h32, n);
    transpose_w_kernel<<<(D_ * D_ + 255) / 256, 256, 0, stream>>>(Wq, WqT, D_, D_);
    transpose_w_kernel<<<(D_ * D_ + 255) / 256, 256, 0, stream>>>(Wk, WkT, D_, D_);
    transpose_w_kernel<<<(D_ * D_ + 255) / 256, 256, 0, stream>>>(Wv, WvT, D_, D_);
    transpose_w_kernel<<<(D_ * HID_ + 255) / 256, 256, 0, stream>>>(W1, W1T, D_, HID_);
    transpose_w_kernel<<<(HID_ * D_ + 255) / 256, 256, 0, stream>>>(W2, W2T, HID_, D_);

    const int M = B_ * S_;
    const int gemmBlocks = (M / 128) * (D_ / 64);   // 64 * 8 = 512
    // k = x @ Wk + bk (row-major); v = x @ Wv + bv stored as V^T [B][D][S]
    gemm_bias_f16<<<gemmBlocks, 256, 0, stream>>>(x16, WkT, bk, k16, M, D_, D_, 0);
    gemm_bias_f16<<<gemmBlocks, 256, 0, stream>>>(x16, WvT, bv, vT,  M, D_, D_, 1);

    const int attnBlocks = B_ * (S_ / 16);          // 512
    for (int layer = 0; layer < L_; ++layer) {
        gemm_bias_f16<<<gemmBlocks, 256, 0, stream>>>(h16, WqT, bq, q16, M, D_, D_, 0);
        float* outp = (layer == L_ - 1) ? (float*)d_out : h32;
        attn_ffn_ln_kernel<<<attnBlocks, 256, 0, stream>>>(
            q16, k16, vT, W1T, b1, W2T, b2,
            ln1g, ln1b, ln2g, ln2b, h32, outp, h16);
    }
}
